// MGRU_55241869361387
// MI455X (gfx1250) — compile-verified, matrix operand order checked by reading, not run on hardware
//
#include <hip/hip_runtime.h>

// ---------------------------------------------------------------------------
// MGRU on gfx1250 (MI455X): f16 WMMA w/ f32 accumulate everywhere.
//   T=512 B=64 I=512 H=1024 O=512 K=64
// Critical-path engineering:
//   * Gx = X@Wx precomputed (parallel GEMM)  -> h-chain K: 1536 -> 1024
//   * Phm = H@Whm precomputed (parallel GEMM)-> m-chain is [64x64]@[64x64]
//   * m-scan: ONE persistent workgroup, Wmm in registers, m state in LDS
//   * output projection batched over all T at the end
// ---------------------------------------------------------------------------

typedef __attribute__((ext_vector_type(16))) _Float16 v16h;
typedef __attribute__((ext_vector_type(8)))  _Float16 v8h;
typedef __attribute__((ext_vector_type(8)))  float    v8f;

constexpr int kT = 512;   // time steps
constexpr int kB = 64;    // batch
constexpr int kI = 512;   // input dim
constexpr int kH = 1024;  // hidden dim
constexpr int kO = 512;   // output dim
constexpr int kK = 64;    // memory dim
constexpr int kG = 3 * kH;    // 3072 gate columns

__device__ __forceinline__ v8f wmma_f16(v16h a, v16h b, v8f c) {
  // v_wmma_f32_16x16x32_f16
  return __builtin_amdgcn_wmma_f32_16x16x32_f16(false, a, false, b,
                                                (short)0, c, false, false);
}

// Load one 16x32 f16 fragment from a row-major [rows, ld] matrix.
// ISA 16-bit A layout: lane L -> row L%16; VGPR0..3 = K[klo..klo+7],
// VGPR4..7 = K[klo+16..klo+23], klo = 8*(L/16).  B fragments use the same
// pattern because weights are stored N-major (pre-transposed).  Works for
// global or LDS pointers (compiler keeps the address space through inlining).
__device__ __forceinline__ v16h load_frag(const _Float16* __restrict__ base,
                                          int row0, int k0, int ld) {
  const int lane = threadIdx.x & 31;
  const _Float16* p =
      base + (size_t)(row0 + (lane & 15)) * ld + k0 + ((lane >> 4) << 3);
  v8h lo = *(const v8h*)(p);
  v8h hi = *(const v8h*)(p + 16);
  v16h f;
#pragma unroll
  for (int i = 0; i < 8; ++i) { f[i] = lo[i]; f[8 + i] = hi[i]; }
  return f;
}

__device__ __forceinline__ float sigmoidf_(float x) {
  return 1.0f / (1.0f + __expf(-x));
}

// ---------------------------- packing kernels ------------------------------

// dst[n][k] = src[k][n] for src [Kd, N] row-major (f32 -> f16 transpose).
__global__ void pack_t(const float* __restrict__ src,
                       _Float16* __restrict__ dst, int Kd, int N) {
  int idx = blockIdx.x * blockDim.x + threadIdx.x;
  if (idx >= Kd * N) return;
  int n = idx / Kd, k = idx % Kd;
  dst[idx] = (_Float16)src[(size_t)k * N + n];
}

__global__ void f32_to_f16(const float* __restrict__ s,
                           _Float16* __restrict__ d, int n) {
  int i = blockIdx.x * blockDim.x + threadIdx.x;
  if (i < n) d[i] = (_Float16)s[i];
}

// --------------------------- parallel GEMMs --------------------------------

// C[M,N] (f16) = A[M,Kd] (row-major f16) @ Bt[N,Kd] (N-major f16).
// Grid (N/64, M/64), block 256: wave (mi, half) owns two 16x16 tiles.
__global__ __launch_bounds__(256) void gemm_nt(
    const _Float16* __restrict__ A, const _Float16* __restrict__ Bt,
    _Float16* __restrict__ C, int Kd, int N) {
  const int wave = threadIdx.x >> 5;
  const int lane = threadIdx.x & 31;
  const int row0 = blockIdx.y * 64 + (wave & 3) * 16;
  const int n0a = blockIdx.x * 64 + (wave >> 2) * 32;
  const int n0b = n0a + 16;
  v8f accA = {}, accB = {};
  for (int k = 0; k < Kd; k += 32) {
    v16h a  = load_frag(A, row0, k, Kd);
    v16h b0 = load_frag(Bt, n0a, k, Kd);
    v16h b1 = load_frag(Bt, n0b, k, Kd);
    accA = wmma_f16(a, b0, accA);
    accB = wmma_f16(a, b1, accB);
  }
  const int rbase = row0 + ((lane >> 4) << 3);
  const int ca = n0a + (lane & 15), cb = n0b + (lane & 15);
#pragma unroll
  for (int v = 0; v < 8; ++v) {
    C[(size_t)(rbase + v) * N + ca] = (_Float16)accA[v];
    C[(size_t)(rbase + v) * N + cb] = (_Float16)accB[v];
  }
}

// ------------------------- recurrent h-scan step ---------------------------

// h_new = (1-z)*n + z*h, with gx = (X@Wx)[t] precomputed.
// Grid: H/32 = 32 WGs, block 256 (8 waves). Wave (mi, ni): one 16x16 tile
// per gate. K-loop: 32 chunks over h only.
__global__ __launch_bounds__(256) void gru_step(
    const _Float16* __restrict__ Gxt,    // [64,3072] this step's x-gates
    const _Float16* __restrict__ Hprev,  // [64,1024]
    const _Float16* __restrict__ Wh_t,   // [3072,1024] N-major
    const float* __restrict__ b,         // [3072]
    _Float16* __restrict__ Hnew)         // [64,1024]
{
  const int wave = threadIdx.x >> 5;
  const int lane = threadIdx.x & 31;
  const int row0 = (wave & 3) * 16;
  const int n0 = blockIdx.x * 32 + (wave >> 2) * 16;  // column within H

  v8f az = {}, ar = {}, an = {};
  for (int k = 0; k < kH; k += 32) {
    v16h a  = load_frag(Hprev, row0, k, kH);
    v16h bz = load_frag(Wh_t, 0 * kH + n0, k, kH);
    v16h br = load_frag(Wh_t, 1 * kH + n0, k, kH);
    v16h bn = load_frag(Wh_t, 2 * kH + n0, k, kH);
    az = wmma_f16(a, bz, az);
    ar = wmma_f16(a, br, ar);
    an = wmma_f16(a, bn, an);
  }

  // C layout: lane L, vgpr v -> row = 8*(L/16)+v, col = L%16.
  const int col = n0 + (lane & 15);
  const int rbase = row0 + ((lane >> 4) << 3);
  const float bz_ = b[col], br_ = b[kH + col], bn_ = b[2 * kH + col];
#pragma unroll
  for (int v = 0; v < 8; ++v) {
    const int r = rbase + v;
    const _Float16* gx = Gxt + (size_t)r * kG;
    float z  = sigmoidf_(az[v] + (float)gx[col] + bz_);
    float rr = sigmoidf_(ar[v] + (float)gx[kH + col] + br_);
    float n  = tanhf((float)gx[2 * kH + col] + bn_ + rr * an[v]);
    float hp = (float)Hprev[(size_t)r * kH + col];
    Hnew[(size_t)r * kH + col] = (_Float16)((1.0f - z) * n + z * hp);
  }
}

// --------------------- persistent single-WG m-scan -------------------------

// m_t = tanh(Phm[t] + m_{t-1} @ Wmm + bm), all 512 steps in ONE workgroup.
// 16 waves, each owns one 16x16 tile of the [64,64] state.  Wmm fragments
// stay in registers for the whole scan; the running m state lives in LDS.
__global__ __launch_bounds__(512) void mem_scan(
    const _Float16* __restrict__ Phm,    // [T*64, 64] = (H @ Whm)
    const _Float16* __restrict__ Wmm_t,  // [64,64] N-major
    const float* __restrict__ bm,
    _Float16* __restrict__ Mall)         // [T+1,64,64]; writes slots 1..T
{
  __shared__ __align__(16) _Float16 mbuf[kB * kK];
  const int wave = threadIdx.x >> 5;
  const int lane = threadIdx.x & 31;
  const int row0 = (wave & 3) * 16;
  const int n0 = (wave >> 2) * 16;
  const int col = n0 + (lane & 15);
  const int rbase = row0 + ((lane >> 4) << 3);

  // m0 = 0
  for (int i = threadIdx.x; i < kB * kK; i += 512) mbuf[i] = (_Float16)0.0f;

  // Wmm B-fragments: resident in VGPRs for all 512 steps.
  const v16h bw0 = load_frag(Wmm_t, n0, 0, kK);
  const v16h bw1 = load_frag(Wmm_t, n0, 32, kK);
  const float bm_ = bm[col];
  __syncthreads();

  for (int t = 0; t < kT; ++t) {
    v16h a0 = load_frag(mbuf, row0, 0, kK);   // ds_load_b128 from LDS
    v16h a1 = load_frag(mbuf, row0, 32, kK);
    v8f acc = {};
    acc = wmma_f16(a0, bw0, acc);
    acc = wmma_f16(a1, bw1, acc);
    __syncthreads();  // all waves done reading mbuf

    const _Float16* ph = Phm + (size_t)t * kB * kK;
    _Float16* mo = Mall + (size_t)(t + 1) * kB * kK;
#pragma unroll
    for (int v = 0; v < 8; ++v) {
      const int idx = (rbase + v) * kK + col;
      float mval = tanhf(acc[v] + (float)ph[idx] + bm_);
      mbuf[idx] = (_Float16)mval;
      mo[idx]   = (_Float16)mval;
    }
    __syncthreads();  // new m visible to all waves
  }
}

// ------------------------ batched output projection ------------------------

// out[t] = h_t @ Who + m_t @ Wmo + bo.  Grid (O/64, T), block 256.
__global__ __launch_bounds__(256) void out_proj(
    const _Float16* __restrict__ Hall,   // [T+1,64,1024], slot t+1 = h_t
    const _Float16* __restrict__ Mall,   // [T+1,64,64]
    const _Float16* __restrict__ Who_t,  // [512,1024] N-major
    const _Float16* __restrict__ Wmo_t,  // [512,64]   N-major
    const float* __restrict__ bo,
    float* __restrict__ out)             // [T,64,512]
{
  const int t = blockIdx.y;
  const int wave = threadIdx.x >> 5;
  const int lane = threadIdx.x & 31;
  const int row0 = (wave & 3) * 16;
  const int n0a = blockIdx.x * 64 + (wave >> 2) * 32;
  const int n0b = n0a + 16;
  const _Float16* H = Hall + (size_t)(t + 1) * kB * kH;
  const _Float16* M = Mall + (size_t)(t + 1) * kB * kK;
  v8f accA = {}, accB = {};
  for (int k = 0; k < kH; k += 32) {
    v16h a  = load_frag(H, row0, k, kH);
    v16h b0 = load_frag(Who_t, n0a, k, kH);
    v16h b1 = load_frag(Who_t, n0b, k, kH);
    accA = wmma_f16(a, b0, accA);
    accB = wmma_f16(a, b1, accB);
  }
  for (int k = 0; k < kK; k += 32) {
    v16h a  = load_frag(M, row0, k, kK);
    v16h b0 = load_frag(Wmo_t, n0a, k, kK);
    v16h b1 = load_frag(Wmo_t, n0b, k, kK);
    accA = wmma_f16(a, b0, accA);
    accB = wmma_f16(a, b1, accB);
  }
  const int rbase = row0 + ((lane >> 4) << 3);
  const int ca = n0a + (lane & 15), cb = n0b + (lane & 15);
  const float boa = bo[ca], bob = bo[cb];
  float* o = out + (size_t)t * kB * kO;
#pragma unroll
  for (int v = 0; v < 8; ++v) {
    o[(size_t)(rbase + v) * kO + ca] = accA[v] + boa;
    o[(size_t)(rbase + v) * kO + cb] = accB[v] + bob;
  }
}

// Final (h, m) state: f16 -> f32 appended after outputs.
__global__ void final_state(const _Float16* __restrict__ Hall,
                            const _Float16* __restrict__ Mall,
                            float* __restrict__ out) {
  int i = blockIdx.x * blockDim.x + threadIdx.x;
  float* oh = out + (size_t)kT * kB * kO;
  if (i < kB * kH) {
    oh[i] = (float)Hall[(size_t)kT * kB * kH + i];
  } else if (i < kB * kH + kB * kK) {
    int j = i - kB * kH;
    oh[kB * kH + j] = (float)Mall[(size_t)kT * kB * kK + j];
  }
}

// ------------------------------- launcher ----------------------------------

extern "C" void kernel_launch(void* const* d_in, const int* in_sizes, int n_in,
                              void* d_out, int out_size, void* d_ws,
                              size_t ws_size, hipStream_t stream) {
  const float* inputs = (const float*)d_in[0];  // [T,B,I]
  const float* Wx  = (const float*)d_in[1];     // [I,3H]
  const float* Wh  = (const float*)d_in[2];     // [H,3H]
  const float* b   = (const float*)d_in[3];     // [3H]
  const float* Whm = (const float*)d_in[4];     // [H,K]
  const float* Wmm = (const float*)d_in[5];     // [K,K]
  const float* bm  = (const float*)d_in[6];     // [K]
  const float* Who = (const float*)d_in[7];     // [H,O]
  const float* Wmo = (const float*)d_in[8];     // [K,O]
  const float* bo  = (const float*)d_in[9];     // [O]
  float* out = (float*)d_out;

  // Workspace carve (256B-aligned buffers).
  size_t off = 0;
  auto carve = [&](size_t bytes) -> void* {
    void* p = (char*)d_ws + off;
    off += (bytes + 255) & ~(size_t)255;
    return p;
  };
  _Float16* Wx_t  = (_Float16*)carve((size_t)kG * kI * 2);            // 3.1 MB
  _Float16* Wh_t  = (_Float16*)carve((size_t)kG * kH * 2);            // 6.3 MB
  _Float16* Whm_t = (_Float16*)carve((size_t)kK * kH * 2);
  _Float16* Wmm_t = (_Float16*)carve((size_t)kK * kK * 2);
  _Float16* Who_t = (_Float16*)carve((size_t)kO * kH * 2);            // 1 MB
  _Float16* Wmo_t = (_Float16*)carve((size_t)kO * kK * 2);
  _Float16* Xh    = (_Float16*)carve((size_t)kT * kB * kI * 2);       // 33.5 MB
  _Float16* Gx    = (_Float16*)carve((size_t)kT * kB * kG * 2);       // 201 MB
  _Float16* Hall  = (_Float16*)carve((size_t)(kT + 1) * kB * kH * 2); // 67 MB
  _Float16* Mall  = (_Float16*)carve((size_t)(kT + 1) * kB * kK * 2); // 4.2 MB
  _Float16* Phm   = (_Float16*)carve((size_t)kT * kB * kK * 2);       // 4.2 MB

  // Pack weights to N-major f16 (L2-resident for the whole scan).
  pack_t<<<(kI * kG + 255) / 256, 256, 0, stream>>>(Wx, Wx_t, kI, kG);
  pack_t<<<(kH * kG + 255) / 256, 256, 0, stream>>>(Wh, Wh_t, kH, kG);
  pack_t<<<(kH * kK + 255) / 256, 256, 0, stream>>>(Whm, Whm_t, kH, kK);
  pack_t<<<(kK * kK + 255) / 256, 256, 0, stream>>>(Wmm, Wmm_t, kK, kK);
  pack_t<<<(kH * kO + 255) / 256, 256, 0, stream>>>(Who, Who_t, kH, kO);
  pack_t<<<(kK * kO + 255) / 256, 256, 0, stream>>>(Wmo, Wmo_t, kK, kO);
  f32_to_f16<<<(kT * kB * kI + 255) / 256, 256, 0, stream>>>(
      inputs, Xh, kT * kB * kI);

  // Gx = X @ Wx for all timesteps (parallel, off the critical path).
  gemm_nt<<<dim3(kG / 64, kT * kB / 64), 256, 0, stream>>>(Xh, Wx_t, Gx,
                                                           kI, kG);

  // h0 = 0 (slot 0).
  hipMemsetAsync(Hall, 0, (size_t)kB * kH * 2, stream);

  // Sequential h recurrence (the only truly serial launches).
  for (int t = 0; t < kT; ++t) {
    gru_step<<<kH / 32, 256, 0, stream>>>(
        Gx + (size_t)t * kB * kG, Hall + (size_t)t * kB * kH, Wh_t, b,
        Hall + (size_t)(t + 1) * kB * kH);
  }

  // Phm = H @ Whm for all timesteps (Hall slots 1..T are contiguous).
  gemm_nt<<<dim3(1, kT * kB / 64), 256, 0, stream>>>(
      Hall + (size_t)kB * kH, Whm_t, Phm, kH, kK);

  // Persistent single-workgroup m-scan (state in LDS, Wmm in VGPRs).
  mem_scan<<<1, 512, 0, stream>>>(Phm, Wmm_t, bm, Mall);

  // Batched output projection + final states.
  out_proj<<<dim3(kO / 64, kT), 256, 0, stream>>>(Hall, Mall, Who_t, Wmo_t,
                                                  bo, out);
  final_state<<<(kB * kH + kB * kK + 255) / 256, 256, 0, stream>>>(Hall, Mall,
                                                                   out);
}
